// SpatialMemoryModule_58806692217153
// MI455X (gfx1250) — matrix operand order repaired and checked by reading, not compiled, not run on hardware
//
#include <hip/hip_runtime.h>
#include <hip/hip_bf16.h>

// SpatialMemoryModule for MI455X (gfx1250, wave32, WMMA).
// All GEMMs via v_wmma_f32_16x16x32_f16. Flash-style two-pass softmax so the
// 65536x1024 similarity matrix is never materialized (saves ~256MB of HBM
// traffic). Memory-bank tiles (Mc, Mm^T) are staged into LDS with CDNA5
// GLOBAL_LOAD_ASYNC_TO_LDS_B128 (ASYNCcnt) double-buffering, so each block
// fetches them once and all 4 waves consume them from LDS.
//
// Workspace requirement: 18,255,872 bytes (~17.4 MB), all regions fully
// written every call (harness poisons ws with 0xAA).

typedef __attribute__((ext_vector_type(16))) _Float16 v16h;
typedef __attribute__((ext_vector_type(8)))  _Float16 v8h;
typedef __attribute__((ext_vector_type(8)))  float    v8f;

#define WAVES   4
#define HWSZ    4096            // H*W
#define CHW     (256*4096)      // C*H*W
#define EPSF    1e-12f
#define SHRINKF 0.0025f

__device__ __forceinline__ v8f wmma16(v16h a, v16h b, v8f c) {
  // D = A(16x32 f16) * B(32x16 f16) + C(16x16 f32)
  return __builtin_amdgcn_wmma_f32_16x16x32_f16(
      /*neg_a=*/false, a, /*neg_b=*/false, b,
      /*c_mod=*/(short)0, c, /*reuse_a=*/false, /*reuse_b=*/false);
}

// A-fragment loader. p = row_base + k0 (f16, K contiguous in memory).
// Lane in half h holds K = h*16+0..15 in elements 0..15 (same order as B).
__device__ __forceinline__ v16h load_afrag(const _Float16* p, int half) {
  return *(const v16h*)(p + half * 16);
}

// --- CDNA5 async global->LDS staging (CDNA5 ISA) -----------
// VGLOBAL encoding, GV mode: VDST = per-lane LDS byte address, VADDR = 64-bit
// global address, SADDR = off. Tracked by ASYNCcnt.
__device__ __forceinline__ void async_copy16(unsigned int lds_addr, const void* gaddr) {
  asm volatile("global_load_async_to_lds_b128 %0, %1, off"
               :: "v"(lds_addr), "v"(gaddr) : "memory");
}
__device__ __forceinline__ void wait_async0() {
  asm volatile("s_wait_asynccnt 0x0" ::: "memory");
}
// Flat address of a __shared__ object: low 32 bits are the LDS byte offset
// (ISA §10.2: LDS aperture truncates to addr[31:0]).
__device__ __forceinline__ unsigned int lds_addr_of(const void* p) {
  return (unsigned int)(size_t)p;
}
// Stage a contiguous block of `bytes` (multiple of 2048) with 128 threads.
__device__ __forceinline__ void stage_block(unsigned int lds_base,
                                            const _Float16* g, int bytes, int tid) {
  for (int o = tid * 16; o < bytes; o += 128 * 16)
    async_copy16(lds_base + o, (const char*)g + o);
}

// ---------------------------------------------------------------------------
// Kernel 0: transpose weights to f16 B-side tables.
// Wq1 [256,128] -> Wq1T [128][256]; Wq2 [128,128] -> Wq2T [128][128];
// Wo  [128,256] -> WoT  [256][128].
__global__ __launch_bounds__(256) void k_wprep(
    const float* __restrict__ Wq1, const float* __restrict__ Wq2,
    const float* __restrict__ Wo,
    _Float16* __restrict__ Wq1T, _Float16* __restrict__ Wq2T,
    _Float16* __restrict__ WoT)
{
  const int idx = blockIdx.x * blockDim.x + threadIdx.x;   // 32768 threads
  if (idx < 256 * 128) { int k = idx >> 7, n = idx & 127; Wq1T[n * 256 + k] = (_Float16)Wq1[idx]; }
  if (idx < 128 * 128) { int k = idx >> 7, n = idx & 127; Wq2T[n * 128 + k] = (_Float16)Wq2[idx]; }
  if (idx < 128 * 256) { int k = idx >> 8, n = idx & 255; WoT [n * 128 + k] = (_Float16)Wo [idx]; }
}

// ---------------------------------------------------------------------------
// Kernel 1: memory bank prep. One block (128 thr) per slot (1024 blocks).
__global__ __launch_bounds__(128) void k_mem(
    const float* __restrict__ Hbuf,
    const float* __restrict__ Wm1, const float* __restrict__ bm1,
    const float* __restrict__ Wm2, const float* __restrict__ bm2,
    _Float16* __restrict__ Mc, _Float16* __restrict__ MmT,
    float* __restrict__ var_m)
{
  __shared__ float sm[128], sh[64], red[128];
  const int tid = threadIdx.x;
  const int slot = blockIdx.x;
  const float v = Hbuf[(size_t)slot * 128 + tid];
  red[tid] = v * v;
  __syncthreads();
  for (int s = 64; s > 0; s >>= 1) { if (tid < s) red[tid] += red[tid + s]; __syncthreads(); }
  const float invn = 1.0f / fmaxf(sqrtf(red[0]), EPSF);
  __syncthreads();
  sm[tid] = v * invn;
  __syncthreads();
  if (tid < 64) {
    float h = bm1[tid];
    for (int k = 0; k < 128; ++k) h += sm[k] * Wm1[k * 64 + tid];
    sh[tid] = fmaxf(h, 0.f);
  }
  __syncthreads();
  float o = bm2[tid];
  for (int k = 0; k < 64; ++k) o += sh[k] * Wm2[k * 128 + tid];
  red[tid] = o;
  __syncthreads();
  for (int s = 64; s > 0; s >>= 1) { if (tid < s) red[tid] += red[tid + s]; __syncthreads(); }
  const float mu = red[0] * (1.0f / 128.0f);
  __syncthreads();
  const float cdev = o - mu;
  red[tid] = cdev * cdev;
  __syncthreads();
  for (int s = 64; s > 0; s >>= 1) { if (tid < s) red[tid] += red[tid + s]; __syncthreads(); }
  const float css = red[0];
  if (tid == 0) var_m[slot] = css * (1.0f / 127.0f);
  const float invc = 1.0f / fmaxf(sqrtf(css), EPSF);
  Mc [(size_t)slot * 128 + tid]  = (_Float16)(cdev * invc);
  MmT[(size_t)tid * 1024 + slot] = (_Float16)o;
}

// ---------------------------------------------------------------------------
// Kernel 2: query MLP. 1024 blocks x 128 thr; each wave owns 16 tokens.
__global__ __launch_bounds__(128) void k_query(
    const float* __restrict__ z,
    const _Float16* __restrict__ Wq1T, const float* __restrict__ bq1,
    const _Float16* __restrict__ Wq2T, const float* __restrict__ bq2,
    _Float16* __restrict__ Zc, float* __restrict__ var_z)
{
  __shared__ __align__(32) char smem[WAVES * 12288];   // per wave: 8KB x/zq + 4KB h
  __shared__ float s_mean[WAVES][16];
  __shared__ float s_inv [WAVES][16];
  const int tid  = threadIdx.x;
  const int wave = tid >> 5, lane = tid & 31;
  const int half = lane >> 4, col = lane & 15;
  const int row0 = blockIdx.x * 64 + wave * 16;
  _Float16* xbuf  = (_Float16*)(smem + wave * 12288);          // 16x256 f16
  _Float16* hbuf  = (_Float16*)(smem + wave * 12288 + 8192);   // 16x128 f16
  float*    zqbuf = (float*)   (smem + wave * 12288);          // 16x128 f32 (reuses xbuf)

  // Phase 1: gather token col (lanes 0-15 = 16 consecutive tokens, two
  // interleaved channel streams per half) -> coalesced 64B segments.
  {
    const int n = row0 + col;
    const size_t base_n = (size_t)(n >> 12) * CHW + (n & (HWSZ - 1));
    float ssq = 0.f;
    for (int c2 = 0; c2 < 256; c2 += 2) {
      const int c = c2 + half;
      const float vv = z[base_n + (size_t)c * HWSZ];
      ssq += vv * vv;
      xbuf[col * 256 + c] = (_Float16)vv;
    }
    ssq += __shfl_xor(ssq, 16, 32);
    if (half == 0) s_inv[wave][col] = 1.0f / fmaxf(sqrtf(ssq), EPSF);
  }
  __builtin_amdgcn_wave_barrier();
  for (int idx = lane; idx < 16 * 256; idx += 32) {
    const int t = idx >> 8;
    xbuf[idx] = (_Float16)((float)xbuf[idx] * s_inv[wave][t]);
  }
  __syncthreads();

  // Phase 2: layer 1 (K=256) -> relu -> hbuf
  v16h af[8];
#pragma unroll
  for (int kc = 0; kc < 8; ++kc) af[kc] = load_afrag(xbuf + col * 256 + kc * 32, half);
#pragma unroll
  for (int j = 0; j < 8; ++j) {
    v8f acc = {};
#pragma unroll
    for (int kc = 0; kc < 8; ++kc) {
      v16h b = *(const v16h*)(Wq1T + (size_t)(j * 16 + col) * 256 + kc * 32 + half * 16);
      acc = wmma16(af[kc], b, acc);
    }
    const float bb = bq1[j * 16 + col];
#pragma unroll
    for (int r = 0; r < 8; ++r)
      hbuf[(r + half * 8) * 128 + j * 16 + col] = (_Float16)fmaxf(acc[r] + bb, 0.f);
  }
  __builtin_amdgcn_wave_barrier();

  // Phase 3: layer 2 (K=128) -> zqbuf (f32, aliases xbuf after sync)
  v16h ah[4];
#pragma unroll
  for (int kc = 0; kc < 4; ++kc) ah[kc] = load_afrag(hbuf + col * 128 + kc * 32, half);
  __syncthreads();
#pragma unroll
  for (int j = 0; j < 8; ++j) {
    v8f acc = {};
#pragma unroll
    for (int kc = 0; kc < 4; ++kc) {
      v16h b = *(const v16h*)(Wq2T + (size_t)(j * 16 + col) * 128 + kc * 32 + half * 16);
      acc = wmma16(ah[kc], b, acc);
    }
    const float bb = bq2[j * 16 + col];
#pragma unroll
    for (int r = 0; r < 8; ++r)
      zqbuf[(r + half * 8) * 128 + j * 16 + col] = acc[r] + bb;
  }
  __syncthreads();

  // Phase 4: per-row mean / unbiased var / centered L2-norm factor
  if (half == 0) {
    float s1 = 0.f, s2 = 0.f;
    for (int d = 0; d < 128; ++d) { const float vv = zqbuf[col * 128 + d]; s1 += vv; s2 += vv * vv; }
    const float mu  = s1 * (1.0f / 128.0f);
    const float css = fmaxf(s2 - 128.0f * mu * mu, 0.f);
    var_z[row0 + col] = css * (1.0f / 127.0f);
    s_mean[wave][col] = mu;
    s_inv [wave][col] = 1.0f / fmaxf(sqrtf(css), EPSF);
  }
  __syncthreads();

  // Phase 5: write Zc (coalesced, 16x128 contiguous tile)
  for (int idx = lane; idx < 16 * 128; idx += 32) {
    const int t = idx >> 7, d = idx & 127;
    Zc[(size_t)(row0 + t) * 128 + d] =
        (_Float16)((zqbuf[t * 128 + d] - s_mean[wave][t]) * s_inv[wave][t]);
  }
}

// ---------------------------------------------------------------------------
// Kernel 3: online softmax stats. Each wave: 16 rows vs all 1024 slots.
// Mc is staged into LDS in 64-slot chunks with async double-buffering; the
// four waves of the block share each staged chunk.
__global__ __launch_bounds__(128) void k_stats(
    const _Float16* __restrict__ Zc, const float* __restrict__ var_z,
    const _Float16* __restrict__ Mc, const float* __restrict__ var_m,
    float* __restrict__ rowmax, float* __restrict__ rowsum)
{
  __shared__ __align__(32) _Float16 s_mc[2][64 * 128];   // 2 x 16KB
  const int tid  = threadIdx.x;
  const int wave = tid >> 5, lane = tid & 31;
  const int half = lane >> 4, col = lane & 15;
  const int row0 = blockIdx.x * 64 + wave * 16;
  v16h af[4];
#pragma unroll
  for (int kc = 0; kc < 4; ++kc)
    af[kc] = load_afrag(Zc + (size_t)(row0 + col) * 128 + kc * 32, half);
  float vz[8], rmax[8], rsum[8];
#pragma unroll
  for (int r = 0; r < 8; ++r) {
    vz[r] = var_z[row0 + r + half * 8]; rmax[r] = -3.0e38f; rsum[r] = 0.f;
  }
  // prologue: stage chunk 0
  stage_block(lds_addr_of(&s_mc[0][0]), Mc, 64 * 128 * 2, tid);
  wait_async0();
  __syncthreads();
  for (int ch = 0; ch < 16; ++ch) {
    const int cur = ch & 1;
    if (ch + 1 < 16)
      stage_block(lds_addr_of(&s_mc[cur ^ 1][0]),
                  Mc + (size_t)(ch + 1) * 64 * 128, 64 * 128 * 2, tid);
    const _Float16* mc = &s_mc[cur][0];
#pragma unroll
    for (int st = 0; st < 4; ++st) {
      const int sl   = st * 16 + col;        // slot within chunk
      const int slot = ch * 64 + sl;
      const float vm = var_m[slot];
      v8f acc = {};
#pragma unroll
      for (int kc = 0; kc < 4; ++kc) {
        v16h b = *(const v16h*)(mc + (size_t)sl * 128 + kc * 32 + half * 16);
        acc = wmma16(af[kc], b, acc);
      }
#pragma unroll
      for (int r = 0; r < 8; ++r) {
        const float s  = acc[r] / (1.0f + fabsf(vz[r] - vm));
        const float nm = fmaxf(rmax[r], s);
        rsum[r] = rsum[r] * __expf(rmax[r] - nm) + __expf(s - nm);
        rmax[r] = nm;
      }
    }
    wait_async0();
    __syncthreads();
  }
  // combine the 16 column-lanes within each half (rows identical per half)
#pragma unroll
  for (int off = 1; off < 16; off <<= 1) {
#pragma unroll
    for (int r = 0; r < 8; ++r) {
      const float om = __shfl_xor(rmax[r], off, 32);
      const float os = __shfl_xor(rsum[r], off, 32);
      const float nm = fmaxf(rmax[r], om);
      rsum[r] = rsum[r] * __expf(rmax[r] - nm) + os * __expf(om - nm);
      rmax[r] = nm;
    }
  }
  if (col == 0) {
#pragma unroll
    for (int r = 0; r < 8; ++r) {
      const int row = row0 + r + half * 8;
      rowmax[row] = rmax[r];
      rowsum[row] = rsum[r];
    }
  }
}

// ---------------------------------------------------------------------------
// Kernel 4: fused weights -> retrieval -> out projection -> NCHW scatter.
// Recomputes sim in 32-slot chunks from async-staged Mc / Mm^T LDS tiles,
// converts to post-shrink weights, feeds them straight into the retrieval
// WMMA; then projects through Wo and scatters to NCHW.
__global__ __launch_bounds__(128) void k_out(
    const _Float16* __restrict__ Zc, const float* __restrict__ var_z,
    const _Float16* __restrict__ Mc, const float* __restrict__ var_m,
    const _Float16* __restrict__ MmT,
    const float* __restrict__ rowmax, const float* __restrict__ rowsum,
    const _Float16* __restrict__ WoT, const float* __restrict__ bo,
    float* __restrict__ out)
{
  __shared__ __align__(32) _Float16 s_mc[2][32 * 128];   // 2 x 8KB  (Mc chunk)
  __shared__ __align__(32) _Float16 s_mt[2][128 * 32];   // 2 x 8KB  ([dim][slot])
  __shared__ __align__(32) _Float16 s_w[WAVES][16 * 32];
  __shared__ __align__(32) _Float16 s_r[WAVES][16 * 128];
  __shared__ float s_o[WAVES][16 * 16];
  const int tid  = threadIdx.x;
  const int wave = tid >> 5, lane = tid & 31;
  const int half = lane >> 4, col = lane & 15;
  const int row0 = blockIdx.x * 64 + wave * 16;
  v16h af[4];
#pragma unroll
  for (int kc = 0; kc < 4; ++kc)
    af[kc] = load_afrag(Zc + (size_t)(row0 + col) * 128 + kc * 32, half);
  float vz[8], rmx[8], rsm[8], tpart[8];
#pragma unroll
  for (int r = 0; r < 8; ++r) {
    const int row = row0 + r + half * 8;
    vz[r] = var_z[row]; rmx[r] = rowmax[row]; rsm[r] = rowsum[row]; tpart[r] = 0.f;
  }
  v8f racc[8] = {};

  // stage Mm^T chunk: 128 rows x 64B gathered into compact [dim][slot] tile
  auto stage_mt = [&](int buf, int cc) {
    const unsigned int base = lds_addr_of(&s_mt[buf][0]);
    for (int u = tid; u < 512; u += 128) {           // 512 x 16B = 8KB
      const int row = u >> 2, q = u & 3;
      async_copy16(base + u * 16,
                   (const char*)(MmT + (size_t)row * 1024 + cc * 32) + q * 16);
    }
  };
  // prologue: stage chunk 0
  stage_block(lds_addr_of(&s_mc[0][0]), Mc, 32 * 128 * 2, tid);
  stage_mt(0, 0);
  wait_async0();
  __syncthreads();

  for (int cc = 0; cc < 32; ++cc) {
    const int cur = cc & 1;
    if (cc + 1 < 32) {
      stage_block(lds_addr_of(&s_mc[cur ^ 1][0]),
                  Mc + (size_t)(cc + 1) * 32 * 128, 32 * 128 * 2, tid);
      stage_mt(cur ^ 1, cc + 1);
    }
    const _Float16* mc = &s_mc[cur][0];
    const _Float16* mt = &s_mt[cur][0];
#pragma unroll
    for (int sub = 0; sub < 2; ++sub) {
      const int sl   = sub * 16 + col;       // slot within chunk
      const int slot = cc * 32 + sl;
      const float vm = var_m[slot];
      v8f acc = {};
#pragma unroll
      for (int kc = 0; kc < 4; ++kc) {
        v16h b = *(const v16h*)(mc + (size_t)sl * 128 + kc * 32 + half * 16);
        acc = wmma16(af[kc], b, acc);
      }
#pragma unroll
      for (int r = 0; r < 8; ++r) {
        const float s = acc[r] / (1.0f + fabsf(vz[r] - vm));
        const float p = __expf(s - rmx[r]);
        const float w = fmaxf(p - SHRINKF * rsm[r], 0.f);   // relu(softmax-c)*sumexp
        tpart[r] += w;
        s_w[wave][(r + half * 8) * 32 + sub * 16 + col] = (_Float16)w;
      }
    }
    __builtin_amdgcn_wave_barrier();
    v16h aw = load_afrag(&s_w[wave][col * 32], half);
#pragma unroll
    for (int j = 0; j < 8; ++j) {
      v16h b = *(const v16h*)(mt + (size_t)(j * 16 + col) * 32 + half * 16);
      racc[j] = wmma16(aw, b, racc[j]);
    }
    wait_async0();
    __syncthreads();
  }
  // per-row total weight (L1 renorm denominator)
#pragma unroll
  for (int off = 1; off < 16; off <<= 1)
#pragma unroll
    for (int r = 0; r < 8; ++r) tpart[r] += __shfl_xor(tpart[r], off, 32);
  float invT[8];
#pragma unroll
  for (int r = 0; r < 8; ++r) invT[r] = 1.0f / fmaxf(tpart[r], EPSF);
#pragma unroll
  for (int j = 0; j < 8; ++j)
#pragma unroll
    for (int r = 0; r < 8; ++r)
      s_r[wave][(r + half * 8) * 128 + j * 16 + col] = (_Float16)(racc[j][r] * invT[r]);
  __builtin_amdgcn_wave_barrier();

  // Output projection: [16x128] @ Wo[128x256] + bo, scatter to NCHW.
  v16h ar[4];
#pragma unroll
  for (int kc = 0; kc < 4; ++kc) ar[kc] = load_afrag(&s_r[wave][col * 128 + kc * 32], half);
  for (int jt = 0; jt < 16; ++jt) {
    v8f acc = {};
#pragma unroll
    for (int kc = 0; kc < 4; ++kc) {
      v16h b = *(const v16h*)(WoT + (size_t)(jt * 16 + col) * 128 + kc * 32 + half * 16);
      acc = wmma16(ar[kc], b, acc);
    }
    const float bb = bo[jt * 16 + col];
#pragma unroll
    for (int r = 0; r < 8; ++r) s_o[wave][(r + half * 8) * 16 + col] = acc[r] + bb;
    __builtin_amdgcn_wave_barrier();
    // transpose through LDS so lanes sweep consecutive tokens (coalesced)
#pragma unroll
    for (int it = 0; it < 8; ++it) {
      const int idx = it * 32 + lane;
      const int t = idx & 15, cl = idx >> 4;
      const int n = row0 + t;
      const int c = jt * 16 + cl;
      out[(size_t)(n >> 12) * CHW + (size_t)c * HWSZ + (n & (HWSZ - 1))] = s_o[wave][t * 16 + cl];
    }
    __builtin_amdgcn_wave_barrier();
  }
}

// ---------------------------------------------------------------------------
extern "C" void kernel_launch(void* const* d_in, const int* in_sizes, int n_in,
                              void* d_out, int out_size, void* d_ws, size_t ws_size,
                              hipStream_t stream)
{
  (void)in_sizes; (void)n_in; (void)out_size; (void)ws_size;
  const float* z   = (const float*)d_in[0];
  const float* Hb  = (const float*)d_in[1];
  const float* Wq1 = (const float*)d_in[2];
  const float* bq1 = (const float*)d_in[3];
  const float* Wq2 = (const float*)d_in[4];
  const float* bq2 = (const float*)d_in[5];
  const float* Wm1 = (const float*)d_in[6];
  const float* bm1 = (const float*)d_in[7];
  const float* Wm2 = (const float*)d_in[8];
  const float* bm2 = (const float*)d_in[9];
  const float* Wo  = (const float*)d_in[10];
  const float* bo  = (const float*)d_in[11];
  float* out = (float*)d_out;
  char* ws = (char*)d_ws;

  _Float16* Wq1T = (_Float16*)(ws + 0);         //  65536 B
  _Float16* Wq2T = (_Float16*)(ws + 65536);     //  32768 B
  _Float16* WoT  = (_Float16*)(ws + 98304);     //  65536 B
  _Float16* Mc   = (_Float16*)(ws + 163840);    // 262144 B
  _Float16* MmT  = (_Float16*)(ws + 425984);    // 262144 B
  float* var_m   = (float*)   (ws + 688128);    //   4096 B
  float* var_z   = (float*)   (ws + 692224);    // 262144 B
  float* rowmax  = (float*)   (ws + 954368);    // 262144 B
  float* rowsum  = (float*)   (ws + 1216512);   // 262144 B
  _Float16* Zc   = (_Float16*)(ws + 1478656);   // 16777216 B  (end: 18255872)

  k_wprep<<<dim3(128),  dim3(256), 0, stream>>>(Wq1, Wq2, Wo, Wq1T, Wq2T, WoT);
  k_mem  <<<dim3(1024), dim3(128), 0, stream>>>(Hb, Wm1, bm1, Wm2, bm2, Mc, MmT, var_m);
  k_query<<<dim3(1024), dim3(128), 0, stream>>>(z, Wq1T, bq1, Wq2T, bq2, Zc, var_z);
  k_stats<<<dim3(1024), dim3(128), 0, stream>>>(Zc, var_z, Mc, var_m, rowmax, rowsum);
  k_out  <<<dim3(1024), dim3(128), 0, stream>>>(Zc, var_z, Mc, var_m, MmT,
                                                rowmax, rowsum, WoT, bo, out);
}
